// RWKV7_DoubleSelfAttBlock_78391743086642
// MI455X (gfx1250) — matrix operand order repaired and checked by reading, not compiled, not run on hardware
//
#include <hip/hip_runtime.h>

// ---------------------------------------------------------------------------
// RWKV-7 double self-attention block for gfx1250 (MI455X).
//  - All dense projections run through v_wmma_f32_16x16x32_bf16 (bf16 in,
//    fp32 accumulate). Weights converted+transposed to bf16 once per launch.
//  - GEMM: 128x64 block tile, 32x32 per wave (4 WMMAs / 4 frag loads),
//    double-buffered LDS, one barrier per K-step, early global loads.
//  - WKV7 scan: one block per (batch,head), 64 threads, state rows in VGPRs.
//  - LN / GN / kk-norm / gating fused into wave32 reduction kernels.
// ---------------------------------------------------------------------------

#define CDIM 1024
#define HEADS 16
#define HDIM 64

typedef __bf16 bf16_t;
typedef bf16_t v16bf __attribute__((ext_vector_type(16)));
typedef float  v8f   __attribute__((ext_vector_type(8)));

static inline int cdiv(int a, int b) { return (a + b - 1) / b; }

__device__ __forceinline__ bf16_t f2bf(float x) {
  union { float f; unsigned u; } in; in.f = x;
  unsigned u = in.u + 0x7FFFu + ((in.u >> 16) & 1u);   // round-to-nearest-even
  union { unsigned short s; bf16_t b; } out; out.s = (unsigned short)(u >> 16);
  return out.b;
}

union FragU { uint4 q[2]; v16bf f; };

// ------------------------------ GEMM ---------------------------------------
// Out[M,N] = A[M,K] (fp32) * W[K,N], with W pre-transposed to Wt[N,K] bf16.
// Block tile 128x64x32; 8 wave32 in a 4x2 grid, each wave owns a 32x32 tile:
// 2 A-frags x 2 B-frags -> 4 v_wmma per K-step. LDS is double buffered so the
// loop has a single barrier per K-step and next-tile global loads are issued
// before the WMMAs. Requires M%128==0, N%64==0, K%32==0 (true for all calls).
__global__ void __launch_bounds__(256)
gemm_bf16_kernel(const float* __restrict__ A, const bf16_t* __restrict__ Wt,
                 float* __restrict__ Out, int N, int K) {
  // 40-halve row stride = 80B = 5*16B: 16B-aligned rows, conflict-free frags
  __shared__ bf16_t lA[2][128][40];
  __shared__ bf16_t lB[2][64][40];
  const int tid  = threadIdx.x;
  const int lane = tid & 31;
  const int wid  = tid >> 5;
  const int mBase = blockIdx.y * 128;
  const int nBase = blockIdx.x * 64;
  const int mw = (wid & 3) * 32;   // wave row group
  const int nw = (wid >> 2) * 32;  // wave col group
  const int arow = tid >> 3;       // 0..31 (4 rows per thread, stride 32)
  const int acol = (tid & 7) * 4;  // 0..28
  const int brow = tid >> 2;       // 0..63
  const int bcol = (tid & 3) * 8;  // 0..24
  const int frow = lane & 15;
  const int akb  = (lane < 16) ? 0 : 8;    // A frag: K {0..7,16..23} vs {8..15,24..31}
  const int bkb  = (lane < 16) ? 0 : 8;    // B frag: K {0..7,16..23} vs {8..15,24..31}

  const float*  pA = A  + (size_t)(mBase + arow) * K + acol;
  const bf16_t* pB = Wt + (size_t)(nBase + brow) * K + bcol;
  const size_t rA = (size_t)32 * K;   // row-group stride for A loads

  float4 fa0, fa1, fa2, fa3;
  uint4  fb;
  // ---- load tile 0 into registers, stage to LDS buffer 0 ----
  fa0 = *(const float4*)(pA);
  fa1 = *(const float4*)(pA + rA);
  fa2 = *(const float4*)(pA + 2 * rA);
  fa3 = *(const float4*)(pA + 3 * rA);
  fb  = *(const uint4*)(pB);
  {
    bf16_t* d0 = &lA[0][arow][acol];
    d0[0] = f2bf(fa0.x); d0[1] = f2bf(fa0.y); d0[2] = f2bf(fa0.z); d0[3] = f2bf(fa0.w);
    bf16_t* d1 = &lA[0][arow + 32][acol];
    d1[0] = f2bf(fa1.x); d1[1] = f2bf(fa1.y); d1[2] = f2bf(fa1.z); d1[3] = f2bf(fa1.w);
    bf16_t* d2 = &lA[0][arow + 64][acol];
    d2[0] = f2bf(fa2.x); d2[1] = f2bf(fa2.y); d2[2] = f2bf(fa2.z); d2[3] = f2bf(fa2.w);
    bf16_t* d3 = &lA[0][arow + 96][acol];
    d3[0] = f2bf(fa3.x); d3[1] = f2bf(fa3.y); d3[2] = f2bf(fa3.z); d3[3] = f2bf(fa3.w);
    *(uint4*)(&lB[0][brow][bcol]) = fb;
  }
  __syncthreads();

  v8f acc00 = {}, acc01 = {}, acc10 = {}, acc11 = {};
  const int nt = K >> 5;
  for (int kt = 0;; ++kt) {
    const int cur = kt & 1;
    if (kt + 1 < nt) {               // issue next tile's global loads early
      const float*  qA = pA + (size_t)(kt + 1) * 32;
      const bf16_t* qB = pB + (size_t)(kt + 1) * 32;
      fa0 = *(const float4*)(qA);
      fa1 = *(const float4*)(qA + rA);
      fa2 = *(const float4*)(qA + 2 * rA);
      fa3 = *(const float4*)(qA + 3 * rA);
      fb  = *(const uint4*)(qB);
      if (kt + 2 < nt)               // near-cache prefetch of the tile after
        __builtin_prefetch(pA + (size_t)(kt + 2) * 32, 0, 3);
    }
    FragU a0, a1, b0, b1;
    a0.q[0] = *(const uint4*)(&lA[cur][mw + frow][akb]);
    a0.q[1] = *(const uint4*)(&lA[cur][mw + frow][akb + 16]);
    a1.q[0] = *(const uint4*)(&lA[cur][mw + 16 + frow][akb]);
    a1.q[1] = *(const uint4*)(&lA[cur][mw + 16 + frow][akb + 16]);
    b0.q[0] = *(const uint4*)(&lB[cur][nw + frow][bkb]);
    b0.q[1] = *(const uint4*)(&lB[cur][nw + frow][bkb + 16]);
    b1.q[0] = *(const uint4*)(&lB[cur][nw + 16 + frow][bkb]);
    b1.q[1] = *(const uint4*)(&lB[cur][nw + 16 + frow][bkb + 16]);
    acc00 = __builtin_amdgcn_wmma_f32_16x16x32_bf16(false, a0.f, false, b0.f,
                                                    (short)0, acc00, false, false);
    acc01 = __builtin_amdgcn_wmma_f32_16x16x32_bf16(false, a0.f, false, b1.f,
                                                    (short)0, acc01, false, false);
    acc10 = __builtin_amdgcn_wmma_f32_16x16x32_bf16(false, a1.f, false, b0.f,
                                                    (short)0, acc10, false, false);
    acc11 = __builtin_amdgcn_wmma_f32_16x16x32_bf16(false, a1.f, false, b1.f,
                                                    (short)0, acc11, false, false);
    if (kt + 1 == nt) break;
    const int nxt = cur ^ 1;         // stage next tile into the other buffer
    bf16_t* d0 = &lA[nxt][arow][acol];
    d0[0] = f2bf(fa0.x); d0[1] = f2bf(fa0.y); d0[2] = f2bf(fa0.z); d0[3] = f2bf(fa0.w);
    bf16_t* d1 = &lA[nxt][arow + 32][acol];
    d1[0] = f2bf(fa1.x); d1[1] = f2bf(fa1.y); d1[2] = f2bf(fa1.z); d1[3] = f2bf(fa1.w);
    bf16_t* d2 = &lA[nxt][arow + 64][acol];
    d2[0] = f2bf(fa2.x); d2[1] = f2bf(fa2.y); d2[2] = f2bf(fa2.z); d2[3] = f2bf(fa2.w);
    bf16_t* d3 = &lA[nxt][arow + 96][acol];
    d3[0] = f2bf(fa3.x); d3[1] = f2bf(fa3.y); d3[2] = f2bf(fa3.z); d3[3] = f2bf(fa3.w);
    *(uint4*)(&lB[nxt][brow][bcol]) = fb;
    __syncthreads();
  }

  const int orow0 = mBase + mw + ((lane < 16) ? 0 : 8);
  const int orow1 = orow0 + 16;
  const int ocol0 = nBase + nw + frow;
#pragma unroll
  for (int j = 0; j < 8; ++j) {
    Out[(size_t)(orow0 + j) * N + ocol0]      = acc00[j];
    Out[(size_t)(orow0 + j) * N + ocol0 + 16] = acc01[j];
    Out[(size_t)(orow1 + j) * N + ocol0]      = acc10[j];
    Out[(size_t)(orow1 + j) * N + ocol0 + 16] = acc11[j];
  }
}

// Wt[n*K + k] = bf16(W[k*N + n])
__global__ void convT_kernel(const float* __restrict__ W, bf16_t* __restrict__ Wt,
                             int K, int N, int total) {
  int i = blockIdx.x * 256 + threadIdx.x;
  if (i >= total) return;
  int n = i / K, k = i - n * K;
  Wt[i] = f2bf(W[(size_t)k * N + n]);
}

// ------------------------------ LayerNorm ----------------------------------
__global__ void __launch_bounds__(256)
ln_kernel(const float* __restrict__ x, const float* __restrict__ g,
          const float* __restrict__ b, float* __restrict__ out, float eps) {
  __shared__ float sred[2][8];
  const size_t base = (size_t)blockIdx.x * CDIM;
  float v[4], s = 0.f, s2 = 0.f;
#pragma unroll
  for (int i = 0; i < 4; ++i) {
    v[i] = x[base + threadIdx.x + i * 256];
    s += v[i]; s2 += v[i] * v[i];
  }
#pragma unroll
  for (int o = 16; o; o >>= 1) { s += __shfl_xor(s, o, 32); s2 += __shfl_xor(s2, o, 32); }
  int wv = threadIdx.x >> 5, lane = threadIdx.x & 31;
  if (!lane) { sred[0][wv] = s; sred[1][wv] = s2; }
  __syncthreads();
  if (threadIdx.x == 0) {
    float a = 0.f, c2 = 0.f;
    for (int i = 0; i < 8; ++i) { a += sred[0][i]; c2 += sred[1][i]; }
    sred[0][0] = a; sred[1][0] = c2;
  }
  __syncthreads();
  const float mean = sred[0][0] * (1.f / CDIM);
  const float var  = sred[1][0] * (1.f / CDIM) - mean * mean;
  const float inv  = rsqrtf(var + eps);
#pragma unroll
  for (int i = 0; i < 4; ++i) {
    int c = threadIdx.x + i * 256;
    out[base + c] = (v[i] - mean) * inv * g[c] + b[c];
  }
}

// ------------------------------ token shift + mix --------------------------
__global__ void shift_mix6_kernel(const float* __restrict__ x,
    const float* __restrict__ m0, const float* __restrict__ m1,
    const float* __restrict__ m2, const float* __restrict__ m3,
    const float* __restrict__ m4, const float* __restrict__ m5,
    float* __restrict__ o0, float* __restrict__ o1, float* __restrict__ o2,
    float* __restrict__ o3, float* __restrict__ o4, float* __restrict__ o5,
    int T, int total) {
  int i = blockIdx.x * 256 + threadIdx.x;
  if (i >= total) return;
  int c = i & (CDIM - 1);
  int t = i / CDIM;
  float xv = x[i];
  float xp = ((t % T) == 0) ? 0.f : x[i - CDIM];
  float xx = xp - xv;
  o0[i] = xv + xx * m0[c]; o1[i] = xv + xx * m1[c]; o2[i] = xv + xx * m2[c];
  o3[i] = xv + xx * m3[c]; o4[i] = xv + xx * m4[c]; o5[i] = xv + xx * m5[c];
}

__global__ void shift_mix1_kernel(const float* __restrict__ x,
    const float* __restrict__ m, float* __restrict__ o, int T, int total) {
  int i = blockIdx.x * 256 + threadIdx.x;
  if (i >= total) return;
  int c = i & (CDIM - 1);
  int t = i / CDIM;
  float xv = x[i];
  float xp = ((t % T) == 0) ? 0.f : x[i - CDIM];
  o[i] = xv + (xp - xv) * m[c];
}

// ------------------------------ elementwise --------------------------------
__global__ void tanh_kernel(float* __restrict__ x, int n) {
  int i = blockIdx.x * 256 + threadIdx.x;
  if (i < n) x[i] = tanhf(x[i]);
}
__global__ void sigmoid_kernel(float* __restrict__ x, int n) {
  int i = blockIdx.x * 256 + threadIdx.x;
  if (i < n) x[i] = 1.f / (1.f + expf(-x[i]));
}
// a = sigmoid(a0 + x)
__global__ void asig_kernel(float* __restrict__ x, const float* __restrict__ a0, int n) {
  int i = blockIdx.x * 256 + threadIdx.x;
  if (i < n) x[i] = 1.f / (1.f + expf(-(a0[i & (CDIM - 1)] + x[i])));
}
// w = -softplus(-(w0+x)) - 0.5 ;  wd = exp(-exp(w))
__global__ void wdecay_kernel(float* __restrict__ x, const float* __restrict__ w0, int n) {
  int i = blockIdx.x * 256 + threadIdx.x;
  if (i >= n) return;
  float z = w0[i & (CDIM - 1)] + x[i];
  float w = -log1pf(expf(-z)) - 0.5f;
  x[i] = expf(-expf(w));
}
__global__ void relusq_kernel(float* __restrict__ x, int n) {
  int i = blockIdx.x * 256 + threadIdx.x;
  if (i < n) { float v = fmaxf(x[i], 0.f); x[i] = v * v; }
}
__global__ void add_inplace_kernel(float* __restrict__ x, const float* __restrict__ y, int n) {
  int i = blockIdx.x * 256 + threadIdx.x;
  if (i < n) x[i] += y[i];
}
// q[b,t] += ao[b, Tqkv-Tq+t]
__global__ void add_tail_kernel(float* __restrict__ q, const float* __restrict__ ao,
                                int Tq, int Tqkv, int total) {
  int i = blockIdx.x * 256 + threadIdx.x;
  if (i >= total) return;
  int c = i & (CDIM - 1);
  int t = (i / CDIM) % Tq;
  int b = (i / CDIM) / Tq;
  q[i] += ao[(((size_t)b * Tqkv) + (Tqkv - Tq) + t) * CDIM + c];
}

// ------------------------------ kk-normalize + k scale ---------------------
// wave32 per (token,head); lane handles channels (lane, lane+32).
__global__ void __launch_bounds__(256)
kvprep_kernel(float* __restrict__ k, const float* __restrict__ a,
              const float* __restrict__ k_k, const float* __restrict__ k_a,
              float* __restrict__ a_in, float* __restrict__ b_in, int nwaves) {
  int gw   = (blockIdx.x * 256 + threadIdx.x) >> 5;
  int lane = threadIdx.x & 31;
  if (gw >= nwaves) return;
  int h = gw % HEADS;
  size_t base = (size_t)(gw / HEADS) * CDIM + (size_t)h * HDIM;
  int c0 = lane, c1 = lane + 32;
  int g0 = h * HDIM + c0, g1 = h * HDIM + c1;
  float k0v = k[base + c0], k1v = k[base + c1];
  float a0v = a[base + c0], a1v = a[base + c1];
  float kk0 = k0v * k_k[g0], kk1 = k1v * k_k[g1];
  float ss = kk0 * kk0 + kk1 * kk1;
#pragma unroll
  for (int o = 16; o; o >>= 1) ss += __shfl_xor(ss, o, 32);
  float inv = 1.f / fmaxf(sqrtf(ss), 1e-12f);
  kk0 *= inv; kk1 *= inv;
  a_in[base + c0] = -kk0;        a_in[base + c1] = -kk1;
  b_in[base + c0] = kk0 * a0v;   b_in[base + c1] = kk1 * a1v;
  k[base + c0] = k0v * (1.f + (a0v - 1.f) * k_a[g0]);
  k[base + c1] = k1v * (1.f + (a1v - 1.f) * k_a[g1]);
}

// ------------------------------ WKV7 scan ----------------------------------
// One block per (batch,head). 64 threads; thread i owns state row S[i][0..63]
// in registers. Per step: sa_i = S·a ; S = S*w + sa·b^T + v·k^T ; y_i = S·r.
__global__ void __launch_bounds__(64)
wkv7_kernel(const float* __restrict__ r, const float* __restrict__ wd,
            const float* __restrict__ k, const float* __restrict__ v,
            const float* __restrict__ a, const float* __restrict__ b,
            float* __restrict__ y, int T) {
  const int bb = blockIdx.x / HEADS, h = blockIdx.x % HEADS;
  const int i = threadIdx.x;
  __shared__ float sr[64], sw[64], sk[64], sv[64], sa[64], sb[64];
  float S[64];
#pragma unroll
  for (int j = 0; j < 64; ++j) S[j] = 0.f;
  size_t off = ((size_t)bb * T) * CDIM + (size_t)h * HDIM;
  for (int t = 0; t < T; ++t, off += CDIM) {
    sr[i] = r[off + i]; sw[i] = wd[off + i]; sk[i] = k[off + i];
    sv[i] = v[off + i]; sa[i] = a[off + i];  sb[i] = b[off + i];
    __syncthreads();
    float sacc = 0.f;
#pragma unroll
    for (int j = 0; j < 64; ++j) sacc += S[j] * sa[j];
    const float vi = sv[i];
    float yi = 0.f;
#pragma unroll
    for (int j = 0; j < 64; ++j) {
      S[j] = S[j] * sw[j] + sacc * sb[j] + vi * sk[j];
      yi += S[j] * sr[j];
    }
    y[off + i] = yi;
    __syncthreads();
  }
}

// ------------------------------ GN + rk*v + gate ---------------------------
// out = (GroupNorm(y)*gn_g+gn_b + (sum_head r*k*r_k)*v) * g
__global__ void __launch_bounds__(256)
postwkv_kernel(const float* __restrict__ y, const float* __restrict__ r,
               const float* __restrict__ k, const float* __restrict__ v,
               const float* __restrict__ g, const float* __restrict__ r_k,
               const float* __restrict__ gng, const float* __restrict__ gnb,
               float* __restrict__ out, int nwaves) {
  int gw   = (blockIdx.x * 256 + threadIdx.x) >> 5;
  int lane = threadIdx.x & 31;
  if (gw >= nwaves) return;
  int h = gw % HEADS;
  size_t base = (size_t)(gw / HEADS) * CDIM + (size_t)h * HDIM;
  int c0 = lane, c1 = lane + 32;
  int g0 = h * HDIM + c0, g1 = h * HDIM + c1;
  float y0 = y[base + c0], y1 = y[base + c1];
  float s = y0 + y1;
#pragma unroll
  for (int o = 16; o; o >>= 1) s += __shfl_xor(s, o, 32);
  float mean = s * (1.f / HDIM);
  float d0 = y0 - mean, d1 = y1 - mean;
  float s2 = d0 * d0 + d1 * d1;
#pragma unroll
  for (int o = 16; o; o >>= 1) s2 += __shfl_xor(s2, o, 32);
  float inv = rsqrtf(s2 * (1.f / HDIM) + 64e-5f);
  float rkp = r[base + c0] * k[base + c0] * r_k[g0]
            + r[base + c1] * k[base + c1] * r_k[g1];
#pragma unroll
  for (int o = 16; o; o >>= 1) rkp += __shfl_xor(rkp, o, 32);
  float n0 = d0 * inv * gng[g0] + gnb[g0];
  float n1 = d1 * inv * gng[g1] + gnb[g1];
  out[base + c0] = (n0 + rkp * v[base + c0]) * g[base + c0];
  out[base + c1] = (n1 + rkp * v[base + c1]) * g[base + c1];
}

// ===========================================================================
// Host orchestration
// ===========================================================================
namespace {

struct AttnP {
  const float *x_r, *x_w, *x_k, *x_v, *x_a, *x_g;
  const float *w0, *a0, *k_k, *k_a, *r_k, *gn_g, *gn_b;
  const bf16_t *w1t, *w2t, *a1t, *a2t, *g1t, *g2t, *Wrt, *Wkt, *Wvt, *Wot;
};

void gemm(const float* A, const bf16_t* Wt, float* Out, int M, int N, int K,
          hipStream_t st) {
  dim3 grid(N / 64, M / 128);
  gemm_bf16_kernel<<<grid, 256, 0, st>>>(A, Wt, Out, N, K);
}

// x is already LayerNorm'ed attn input. Uses 7 NT*C work slots + tmp(NT*128).
void run_attn(const float* xln, int B, int T, const AttnP& P, float* vbuf,
              float* outbuf, float* const* w, float* tmp, hipStream_t st) {
  const int NTOK = B * T;
  const int tot = NTOK * CDIM;
  const int eg = cdiv(tot, 256);
  shift_mix6_kernel<<<eg, 256, 0, st>>>(xln, P.x_r, P.x_w, P.x_k, P.x_v, P.x_a,
                                        P.x_g, w[0], w[1], w[2], w[3], w[4], w[5],
                                        T, tot);
  gemm(w[0], P.Wrt, w[6], NTOK, CDIM, CDIM, st);                    // r -> w6
  gemm(w[1], P.w1t, tmp, NTOK, 64, CDIM, st);                        // xw@w1
  tanh_kernel<<<cdiv(NTOK * 64, 256), 256, 0, st>>>(tmp, NTOK * 64);
  gemm(tmp, P.w2t, w[0], NTOK, CDIM, 64, st);                        // ..@w2 -> w0
  wdecay_kernel<<<eg, 256, 0, st>>>(w[0], P.w0, tot);                // wd -> w0
  gemm(w[2], P.Wkt, w[1], NTOK, CDIM, CDIM, st);                     // k -> w1
  gemm(w[3], P.Wvt, vbuf, NTOK, CDIM, CDIM, st);                     // v -> vbuf
  gemm(w[4], P.a1t, tmp, NTOK, 64, CDIM, st);                        // xa@a1
  gemm(tmp, P.a2t, w[2], NTOK, CDIM, 64, st);                        // ..@a2 -> w2
  asig_kernel<<<eg, 256, 0, st>>>(w[2], P.a0, tot);                  // a -> w2
  gemm(w[5], P.g1t, tmp, NTOK, 128, CDIM, st);                       // xg@g1
  sigmoid_kernel<<<cdiv(NTOK * 128, 256), 256, 0, st>>>(tmp, NTOK * 128);
  gemm(tmp, P.g2t, w[3], NTOK, CDIM, 128, st);                       // g -> w3
  const int nwaves = NTOK * HEADS;
  kvprep_kernel<<<cdiv(nwaves * 32, 256), 256, 0, st>>>(w[1], w[2], P.k_k, P.k_a,
                                                        w[4], w[5], nwaves);
  wkv7_kernel<<<B * HEADS, 64, 0, st>>>(w[6], w[0], w[1], vbuf, w[4], w[5],
                                        w[2], T);                     // y -> w2
  postwkv_kernel<<<cdiv(nwaves * 32, 256), 256, 0, st>>>(w[2], w[6], w[1], vbuf,
                                                         w[3], P.r_k, P.gn_g,
                                                         P.gn_b, w[4], nwaves);
  gemm(w[4], P.Wot, outbuf, NTOK, CDIM, CDIM, st);                   // out proj
}

void run_ffn(float* x, int NTOK, int T, const float* fng, const float* fnb,
             const float* xkmix, const bf16_t* Wkt, const bf16_t* Wvt,
             float* s0, float* s1, float* s2, float* s3, hipStream_t st) {
  const int tot = NTOK * CDIM;
  ln_kernel<<<NTOK, 256, 0, st>>>(x, fng, fnb, s0, 1e-5f);
  shift_mix1_kernel<<<cdiv(tot, 256), 256, 0, st>>>(s0, xkmix, s1, T, tot);
  gemm(s1, Wkt, s2, NTOK, 4 * CDIM, CDIM, st);
  relusq_kernel<<<cdiv(NTOK * 4 * CDIM, 256), 256, 0, st>>>(s2, NTOK * 4 * CDIM);
  gemm(s2, Wvt, s3, NTOK, CDIM, 4 * CDIM, st);
  add_inplace_kernel<<<cdiv(tot, 256), 256, 0, st>>>(x, s3, tot);
}

// Flattened input indices: recursive dict insertion order of setup_inputs().
enum {
  IN_QUERY = 0, IN_KEYVAL = 1, IN_QVF = 2, IN_QKVVF = 3,
  QA_PRE_G = 4, QA_PRE_B = 5, QA_AN_G = 6, QA_AN_B = 7, QA_FN_G = 8, QA_FN_B = 9,
  QA_ATT = 10,   // x_r..gn_b (23 entries)
  QA_FFN_XK = 33, QA_FFN_WK = 34, QA_FFN_WV = 35,
  TOP_PRE_G = 36, TOP_PRE_B = 37, TOP_QN_G = 38, TOP_QN_B = 39,
  TOP_FN_G = 40, TOP_FN_B = 41,
  QKV_ATT = 42,
  TOP_FFN_XK = 65, TOP_FFN_WK = 66, TOP_FFN_WV = 67,
};

}  // namespace

extern "C" void kernel_launch(void* const* d_in, const int* in_sizes, int n_in,
                              void* d_out, int out_size, void* d_ws, size_t ws_size,
                              hipStream_t stream) {
  (void)in_sizes; (void)n_in; (void)out_size; (void)ws_size;
  const int B = 4, Tq = 1024, Tk = 3072, Tqkv = 4096;
  const int NQ = B * Tq;        // 4096 query tokens
  const int NQKV = B * Tqkv;    // 16384 qkv tokens
  const size_t SLOT = (size_t)NQKV * CDIM;  // 16M floats per slot

  auto F = [&](int i) { return (const float*)d_in[i]; };

  // ---- workspace layout: [bf16 weights][tmp NT*128 fp32][13 x NT*C fp32] ----
  uint8_t* p = (uint8_t*)d_ws;
  bf16_t* wb = (bf16_t*)p;
  const size_t WB_ELEMS = 26214400;  // exact total of converted weights
  p += (WB_ELEMS * sizeof(bf16_t) + 255) & ~(size_t)255;
  float* tmp = (float*)p;  p += (size_t)NQKV * 128 * sizeof(float);
  float* slot[13];
  for (int i = 0; i < 13; ++i) { slot[i] = (float*)p; p += SLOT * sizeof(float); }

  // ---- convert + transpose weights to bf16 (N x K layout) ----
  bf16_t* cur = wb;
  auto conv = [&](int idx, int K, int N) -> bf16_t* {
    bf16_t* dst = cur; cur += (size_t)K * N;
    int total = K * N;
    convT_kernel<<<cdiv(total, 256), 256, 0, stream>>>(F(idx), dst, K, N, total);
    return dst;
  };
  auto mkAttn = [&](int ib) -> AttnP {
    AttnP P;
    P.x_r = F(ib + 0);  P.x_w = F(ib + 1);  P.x_k = F(ib + 2);
    P.x_v = F(ib + 3);  P.x_a = F(ib + 4);  P.x_g = F(ib + 5);
    P.w0  = F(ib + 6);  P.a0  = F(ib + 9);
    P.k_k = F(ib + 14); P.k_a = F(ib + 15); P.r_k = F(ib + 16);
    P.gn_g = F(ib + 21); P.gn_b = F(ib + 22);
    P.w1t = conv(ib + 7, CDIM, 64);    P.w2t = conv(ib + 8, 64, CDIM);
    P.a1t = conv(ib + 10, CDIM, 64);   P.a2t = conv(ib + 11, 64, CDIM);
    P.g1t = conv(ib + 12, CDIM, 128);  P.g2t = conv(ib + 13, 128, CDIM);
    P.Wrt = conv(ib + 17, CDIM, CDIM); P.Wkt = conv(ib + 18, CDIM, CDIM);
    P.Wvt = conv(ib + 19, CDIM, CDIM); P.Wot = conv(ib + 20, CDIM, CDIM);
    return P;
  };
  AttnP PA = mkAttn(QA_ATT);
  AttnP PB = mkAttn(QKV_ATT);
  bf16_t* ffnA_k = conv(QA_FFN_WK, CDIM, 4 * CDIM);
  bf16_t* ffnA_v = conv(QA_FFN_WV, 4 * CDIM, CDIM);
  bf16_t* ffnB_k = conv(TOP_FFN_WK, CDIM, 4 * CDIM);
  bf16_t* ffnB_v = conv(TOP_FFN_WV, 4 * CDIM, CDIM);

  float* Xq   = slot[0];
  float* xln  = slot[1];
  float* w7[7] = { slot[2], slot[3], slot[4], slot[5], slot[6], slot[7], slot[8] };
  float* vtmp = slot[9];
  float* ao   = slot[10];
  float* Q    = slot[11];
  float* QKV  = slot[12];

  // ---- Stage A: first inner block on query ----
  ln_kernel<<<NQ, 256, 0, stream>>>(F(IN_QUERY), F(QA_PRE_G), F(QA_PRE_B), Xq, 1e-5f);
  ln_kernel<<<NQ, 256, 0, stream>>>(Xq, F(QA_AN_G), F(QA_AN_B), xln, 1e-5f);
  run_attn(xln, B, Tq, PA, vtmp, ao, w7, tmp, stream);
  add_inplace_kernel<<<cdiv(NQ * CDIM, 256), 256, 0, stream>>>(Xq, ao, NQ * CDIM);
  run_ffn(Xq, NQ, Tq, F(QA_FN_G), F(QA_FN_B), F(QA_FFN_XK), ffnA_k, ffnA_v,
          slot[1], slot[2], slot[3], slot[4], stream);

  // ---- Stage B: q = LN(x) ----
  ln_kernel<<<NQ, 256, 0, stream>>>(Xq, F(TOP_PRE_G), F(TOP_PRE_B), Q, 1e-5f);

  // ---- Stage C: qkv = concat([keyval, q], axis=1) per batch ----
  for (int b = 0; b < B; ++b) {
    hipMemcpyAsync(QKV + ((size_t)b * Tqkv) * CDIM,
                   F(IN_KEYVAL) + (size_t)b * Tk * CDIM,
                   (size_t)Tk * CDIM * sizeof(float), hipMemcpyDeviceToDevice, stream);
    hipMemcpyAsync(QKV + ((size_t)b * Tqkv + Tk) * CDIM,
                   Q + (size_t)b * Tq * CDIM,
                   (size_t)Tq * CDIM * sizeof(float), hipMemcpyDeviceToDevice, stream);
  }

  // ---- Stage D: qkv attention; v (=v_first) GEMM'd straight into d_out ----
  float* out_f = (float*)d_out;
  float* qvf_out = out_f + (size_t)NQ * CDIM;
  ln_kernel<<<NQKV, 256, 0, stream>>>(QKV, F(TOP_QN_G), F(TOP_QN_B), xln, 1e-5f);
  run_attn(xln, B, Tqkv, PB, qvf_out, ao, w7, tmp, stream);
  add_tail_kernel<<<cdiv(NQ * CDIM, 256), 256, 0, stream>>>(Q, ao, Tq, Tqkv,
                                                            NQ * CDIM);

  // ---- Stage E: final FFN on q ----
  run_ffn(Q, NQ, Tq, F(TOP_FN_G), F(TOP_FN_B), F(TOP_FFN_XK), ffnB_k, ffnB_v,
          slot[1], slot[2], slot[3], slot[4], stream);

  // ---- Stage F: assemble outputs (q, qvf already written, qkv_v_first) ----
  hipMemcpyAsync(out_f, Q, (size_t)NQ * CDIM * sizeof(float),
                 hipMemcpyDeviceToDevice, stream);
  hipMemcpyAsync(out_f + (size_t)NQ * CDIM + (size_t)NQKV * CDIM, d_in[IN_QKVVF],
                 (size_t)NQKV * CDIM * sizeof(float), hipMemcpyDeviceToDevice,
                 stream);
}
